// MultiheadAttentionModule_59167469470418
// MI455X (gfx1250) — matrix operand order repaired and checked by reading, not compiled, not run on hardware
//
#include <hip/hip_runtime.h>

#define BATCH 2
#define SEQ   4096
#define EMB   768
#define NHEAD 12
#define HDIM  64
#define BN    (BATCH * SEQ)   // 8192
#define APAD  776             // LDS A-tile row stride (bf16): 388 dwords, 388%64=4 -> conflict-free
#define PPAD  72              // LDS P-tile row stride (bf16): 36 dwords, 16 lanes -> 16 distinct banks
#define SCL   0.1803368801111204f   // 0.125 * log2(e): softmax in base-2 domain

typedef __bf16 bf16;
typedef __attribute__((ext_vector_type(16))) __bf16 v16bf;
typedef __attribute__((ext_vector_type(8)))  __bf16 v8bf;
typedef __attribute__((ext_vector_type(8)))  float  v8f;

union FragU { v16bf v; v8bf h[2]; };

// A-matrix 16x32 bf16 fragment (ISA 7.12.2): half 0 -> K = k0+0..7 & k0+16..23,
// half 1 -> K = k0+8..15 & k0+24..31. rowbase = &A[row][0].
__device__ __forceinline__ v16bf load_frag_a(const bf16* rowbase, int k0, int half) {
  FragU f;
  f.h[0] = *(const v8bf*)(rowbase + k0 + half * 8);
  f.h[1] = *(const v8bf*)(rowbase + k0 + 16 + half * 8);
  return f.v;
}

// B-matrix 32x16 bf16 fragment: half 0 -> K = k0+0..7 & k0+16..23, half 1 -> K = k0+8..15 & k0+24..31.
// colbase = &B^T[col][0] (column of B stored contiguously).
__device__ __forceinline__ v16bf load_frag_b(const bf16* colbase, int k0, int half) {
  FragU f;
  f.h[0] = *(const v8bf*)(colbase + k0 + half * 8);
  f.h[1] = *(const v8bf*)(colbase + k0 + 16 + half * 8);
  return f.v;
}

__device__ __forceinline__ v8f wmma_bf16(v16bf a, v16bf b, v8f c) {
  return __builtin_amdgcn_wmma_f32_16x16x32_bf16(
      false, a, false, b, (short)0, c, false, false);
}

// Stage a 16x768 bf16 A tile into LDS (padded rows) with async global->LDS DMA.
// 1536 16B chunks spread over 128 threads (12 each). Caller must __syncthreads().
__device__ __forceinline__ void stage_a_tile_async(bf16* sh, const bf16* g, int tid) {
#pragma unroll
  for (int t = 0; t < 12; ++t) {
    int idx = tid + t * 128;
    int row = idx / 96;          // 0..15
    int ch  = idx % 96;          // 16B chunk within row
    unsigned ldsoff = (unsigned)(size_t)(sh + row * APAD + ch * 8);
    unsigned long long ga = (unsigned long long)(size_t)(g + (size_t)row * EMB + ch * 8);
    asm volatile("global_load_async_to_lds_b128 %0, %1, off"
                 :: "v"(ldsoff), "v"(ga) : "memory");
  }
  asm volatile("s_wait_asynccnt 0x0" ::: "memory");
}

__device__ __forceinline__ v16bf lds_frag_a(const bf16* sh, int row, int k0, int half) {
  FragU f;
  const bf16* rb = sh + row * APAD;
  f.h[0] = *(const v8bf*)(rb + k0 + half * 8);
  f.h[1] = *(const v8bf*)(rb + k0 + 16 + half * 8);
  return f.v;
}

// ---------------------------------------------------------------- prep kernels

__global__ void k_convert_bf16(const float* __restrict__ src, bf16* __restrict__ dst, int n) {
  int i = blockIdx.x * blockDim.x + threadIdx.x;
  if (i < n) dst[i] = (bf16)src[i];
}

// src: [in=768][out=768] f32 -> dst: [out=768][in=768] bf16
__global__ void k_transpose_bf16(const float* __restrict__ src, bf16* __restrict__ dst) {
  int i = blockIdx.x * blockDim.x + threadIdx.x;
  if (i >= EMB * EMB) return;
  int o  = i / EMB;
  int ii = i % EMB;
  dst[i] = (bf16)src[ii * EMB + o];
}

// ---------------------------------------------------------------- QKV GEMM
// One wave -> 16x64 tile; the block's 4 waves share one async-staged A tile.
// mat: 0=Q, 1=K row-major [BN][768]; 2=V stored transposed [B][H][64][4096].
__global__ void k_qkv_gemm(const bf16* __restrict__ xb,
                           const bf16* __restrict__ WqT,
                           const bf16* __restrict__ WkT,
                           const bf16* __restrict__ WvT,
                           bf16* __restrict__ Q,
                           bf16* __restrict__ K,
                           bf16* __restrict__ VT) {
  __shared__ bf16 atile[16 * APAD];

  const int wid  = threadIdx.x >> 5;
  const int lane = threadIdx.x & 31;
  const int half = lane >> 4;
  const int ln   = lane & 15;
  const int w    = blockIdx.x * (blockDim.x >> 5) + wid;

  const int mat   = w / (512 * 12);
  int       rem   = w % (512 * 12);
  const int mtile = rem / 12;
  const int ntile = rem % 12;
  const int m0    = mtile * 16;
  const int n0    = ntile * 64;

  const bf16* WT = (mat == 0) ? WqT : (mat == 1) ? WkT : WvT;

  stage_a_tile_async(atile, xb + (size_t)m0 * EMB, threadIdx.x);
  __syncthreads();

  const bf16* bc[4];
#pragma unroll
  for (int s = 0; s < 4; ++s) bc[s] = WT + (size_t)(n0 + s * 16 + ln) * EMB;

  v8f zero = {};
  v8f acc[4] = {zero, zero, zero, zero};

  // ping-pong double buffering: no rotation copies
  v16bf b0[4], b1[4];
#pragma unroll
  for (int s = 0; s < 4; ++s) b0[s] = load_frag_b(bc[s], 0, half);

  for (int k0 = 0; k0 < EMB; k0 += 64) {
#pragma unroll
    for (int s = 0; s < 4; ++s) b1[s] = load_frag_b(bc[s], k0 + 32, half);
    v16bf a0 = lds_frag_a(atile, ln, k0, half);
#pragma unroll
    for (int s = 0; s < 4; ++s) acc[s] = wmma_bf16(a0, b0[s], acc[s]);

    if (k0 + 64 < EMB) {
#pragma unroll
      for (int s = 0; s < 4; ++s) b0[s] = load_frag_b(bc[s], k0 + 64, half);
    }
    v16bf a1 = lds_frag_a(atile, ln, k0 + 32, half);
#pragma unroll
    for (int s = 0; s < 4; ++s) acc[s] = wmma_bf16(a1, b1[s], acc[s]);
  }

  if (mat < 2) {
    bf16* out = (mat == 0) ? Q : K;
#pragma unroll
    for (int s = 0; s < 4; ++s) {
      int col = n0 + s * 16 + ln;
#pragma unroll
      for (int r = 0; r < 8; ++r) {
        int row = m0 + half * 8 + r;   // C layout: lanes 0-15 rows 0-7, 16-31 rows 8-15
        out[(size_t)row * EMB + col] = (bf16)acc[s][r];
      }
    }
  } else {
    const int b    = m0 >> 12;
    const int tok0 = (m0 & 4095) + half * 8;
#pragma unroll
    for (int s = 0; s < 4; ++s) {
      int c  = n0 + s * 16 + ln;
      int hh = c >> 6, dd = c & 63;
      v8bf pk;
#pragma unroll
      for (int r = 0; r < 8; ++r) pk[r] = (bf16)acc[s][r];
      *(v8bf*)(VT + ((size_t)((b * NHEAD + hh) * HDIM + dd)) * SEQ + tok0) = pk;
    }
  }
}

// ---------------------------------------------------------------- flash attention
// One wave per (b, h, 16-row query tile); 64 keys per iteration (16 WMMAs/iter).
__global__ void k_attention(const bf16* __restrict__ Q,
                            const bf16* __restrict__ K,
                            const bf16* __restrict__ VT,
                            bf16* __restrict__ ctxb) {
  __shared__ bf16 pbuf[4][16 * PPAD];   // per-wave P tile, padded rows

  const int wid  = threadIdx.x >> 5;
  const int lane = threadIdx.x & 31;
  const int half = lane >> 4;
  const int ln   = lane & 15;
  const int w    = blockIdx.x * (blockDim.x >> 5) + wid;

  const int b  = w / (NHEAD * (SEQ / 16));
  int       rm = w % (NHEAD * (SEQ / 16));
  const int h  = rm / (SEQ / 16);
  const int q0 = (rm % (SEQ / 16)) * 16;

  const bf16* Qh = Q + (size_t)(b * SEQ) * EMB + h * HDIM;
  const bf16* Kh = K + (size_t)(b * SEQ) * EMB + h * HDIM;
  const bf16* Vh = VT + (size_t)((b * NHEAD + h) * HDIM) * SEQ;

  const bf16* qrow = Qh + (size_t)(q0 + ln) * EMB;
  v16bf qf0 = load_frag_a(qrow, 0, half);
  v16bf qf1 = load_frag_a(qrow, 32, half);

  v8f zero = {};
  v8f acc[4] = {zero, zero, zero, zero};
  float m[8], l[8];
#pragma unroll
  for (int r = 0; r < 8; ++r) { m[r] = -3.0e38f; l[r] = 0.0f; }

  bf16* pb = &pbuf[wid][0];

  for (int j0 = 0; j0 < SEQ; j0 += 64) {
    // ---- S = Q K^T for a 16x64 key slab (4 C tiles, 8 WMMAs)
    v8f st[4];
#pragma unroll
    for (int t = 0; t < 4; ++t) {
      const bf16* kb = Kh + (size_t)(j0 + t * 16 + ln) * EMB;
      v8f s = zero;
      s = wmma_bf16(qf0, load_frag_b(kb, 0, half), s);
      s = wmma_bf16(qf1, load_frag_b(kb, 32, half), s);
      st[t] = s;
    }

    // ---- issue first V half-slab loads now; softmax VALU hides them
    v16bf vf0[4];
#pragma unroll
    for (int s = 0; s < 4; ++s)
      vf0[s] = load_frag_b(Vh + (size_t)(s * 16 + ln) * SEQ, j0, half);

    if (j0 + 64 < SEQ) {  // warm caches for the next slab (global_prefetch_b8)
#pragma unroll
      for (int t = 0; t < 4; ++t)
        __builtin_prefetch(Kh + (size_t)(j0 + 64 + t * 16 + ln) * EMB, 0, 1);
      __builtin_prefetch(Vh + (size_t)ln * SEQ + j0 + 64, 0, 1);
    }

    // ---- online softmax in base-2 domain (exp2, scale pre-folded).
    // C-layout: element r <-> row (half*8 + r); one row's 16 cols live in a
    // 16-lane half -> shfl_xor 1,2,4,8.
    float corr[8];
#pragma unroll
    for (int r = 0; r < 8; ++r) {
      float x0 = st[0][r] * SCL;
      float x1 = st[1][r] * SCL;
      float x2 = st[2][r] * SCL;
      float x3 = st[3][r] * SCL;
      float bm = fmaxf(fmaxf(x0, x1), fmaxf(x2, x3));
      bm = fmaxf(bm, __shfl_xor(bm, 1, 32));
      bm = fmaxf(bm, __shfl_xor(bm, 2, 32));
      bm = fmaxf(bm, __shfl_xor(bm, 4, 32));
      bm = fmaxf(bm, __shfl_xor(bm, 8, 32));
      float mnew = fmaxf(m[r], bm);
      float cr = exp2f(m[r] - mnew);
      float p0 = exp2f(x0 - mnew);
      float p1 = exp2f(x1 - mnew);
      float p2 = exp2f(x2 - mnew);
      float p3 = exp2f(x3 - mnew);
      float bs = (p0 + p1) + (p2 + p3);
      bs += __shfl_xor(bs, 1, 32);
      bs += __shfl_xor(bs, 2, 32);
      bs += __shfl_xor(bs, 4, 32);
      bs += __shfl_xor(bs, 8, 32);
      l[r] = l[r] * cr + bs;
      m[r] = mnew;
      corr[r] = cr;
      bf16* prow = pb + (half * 8 + r) * PPAD + ln;
      prow[0]  = (bf16)p0;
      prow[16] = (bf16)p1;
      prow[32] = (bf16)p2;
      prow[48] = (bf16)p3;
    }
#pragma unroll
    for (int s = 0; s < 4; ++s)
#pragma unroll
      for (int r = 0; r < 8; ++r) acc[s][r] *= corr[r];

    // ---- reload P as two 16x32 A fragments (same-wave LDS is in-order)
    FragU pf0, pf1;
    pf0.h[0] = *(const v8bf*)(pb + ln * PPAD + half * 8);
    pf0.h[1] = *(const v8bf*)(pb + ln * PPAD + 16 + half * 8);
    pf1.h[0] = *(const v8bf*)(pb + ln * PPAD + 32 + half * 8);
    pf1.h[1] = *(const v8bf*)(pb + ln * PPAD + 48 + half * 8);

    // ---- O += P @ V (second V half-slab loads hidden under first WMMA group)
#pragma unroll
    for (int s = 0; s < 4; ++s) acc[s] = wmma_bf16(pf0.v, vf0[s], acc[s]);

    v16bf vf1[4];
#pragma unroll
    for (int s = 0; s < 4; ++s)
      vf1[s] = load_frag_b(Vh + (size_t)(s * 16 + ln) * SEQ, j0 + 32, half);
#pragma unroll
    for (int s = 0; s < 4; ++s) acc[s] = wmma_bf16(pf1.v, vf1[s], acc[s]);
  }

  // ---- normalize, emit ctx (bf16 row-major [BN][768], feature = h*64+d)
  float inv[8];
#pragma unroll
  for (int r = 0; r < 8; ++r) inv[r] = 1.0f / l[r];
#pragma unroll
  for (int s = 0; s < 4; ++s) {
    int col = h * HDIM + s * 16 + ln;
#pragma unroll
    for (int r = 0; r < 8; ++r) {
      int row = b * SEQ + q0 + half * 8 + r;
      ctxb[(size_t)row * EMB + col] = (bf16)(acc[s][r] * inv[r]);
    }
  }
}

// ---------------------------------------------------------------- output proj
__global__ void k_out_gemm(const bf16* __restrict__ ctxb,
                           const bf16* __restrict__ WoT,
                           const float* __restrict__ bo,
                           float* __restrict__ out) {
  __shared__ bf16 atile[16 * APAD];

  const int wid  = threadIdx.x >> 5;
  const int lane = threadIdx.x & 31;
  const int half = lane >> 4;
  const int ln   = lane & 15;
  const int w    = blockIdx.x * (blockDim.x >> 5) + wid;

  const int mtile = w / 12;
  const int ntile = w % 12;
  const int m0 = mtile * 16;
  const int n0 = ntile * 64;

  stage_a_tile_async(atile, ctxb + (size_t)m0 * EMB, threadIdx.x);
  __syncthreads();

  const bf16* bc[4];
#pragma unroll
  for (int s = 0; s < 4; ++s) bc[s] = WoT + (size_t)(n0 + s * 16 + ln) * EMB;

  v8f zero = {};
  v8f acc[4] = {zero, zero, zero, zero};

  v16bf b0[4], b1[4];
#pragma unroll
  for (int s = 0; s < 4; ++s) b0[s] = load_frag_b(bc[s], 0, half);

  for (int k0 = 0; k0 < EMB; k0 += 64) {
#pragma unroll
    for (int s = 0; s < 4; ++s) b1[s] = load_frag_b(bc[s], k0 + 32, half);
    v16bf a0 = lds_frag_a(atile, ln, k0, half);
#pragma unroll
    for (int s = 0; s < 4; ++s) acc[s] = wmma_bf16(a0, b0[s], acc[s]);

    if (k0 + 64 < EMB) {
#pragma unroll
      for (int s = 0; s < 4; ++s) b0[s] = load_frag_b(bc[s], k0 + 64, half);
    }
    v16bf a1 = lds_frag_a(atile, ln, k0 + 32, half);
#pragma unroll
    for (int s = 0; s < 4; ++s) acc[s] = wmma_bf16(a1, b1[s], acc[s]);
  }

#pragma unroll
  for (int s = 0; s < 4; ++s) {
    int col = n0 + s * 16 + ln;
    float bias = bo[col];
#pragma unroll
    for (int r = 0; r < 8; ++r) {
      int row = m0 + half * 8 + r;
      out[(size_t)row * EMB + col] = acc[s][r] + bias;
    }
  }
}

// ---------------------------------------------------------------- launch

extern "C" void kernel_launch(void* const* d_in, const int* in_sizes, int n_in,
                              void* d_out, int out_size, void* d_ws, size_t ws_size,
                              hipStream_t stream) {
  const float* x  = (const float*)d_in[0];
  const float* Wq = (const float*)d_in[1];
  const float* Wk = (const float*)d_in[2];
  const float* Wv = (const float*)d_in[3];
  const float* Wo = (const float*)d_in[4];
  const float* bo = (const float*)d_in[5];
  float* out = (float*)d_out;

  char* ws = (char*)d_ws;
  const size_t SZ_X = (size_t)BN * EMB * sizeof(bf16);   // 12.6 MB
  const size_t SZ_W = (size_t)EMB * EMB * sizeof(bf16);  // 1.18 MB

  bf16* xb  = (bf16*)(ws);                         // reused as ctx after QKV
  bf16* WqT = (bf16*)(ws + SZ_X);
  bf16* WkT = (bf16*)(ws + SZ_X + SZ_W);
  bf16* WvT = (bf16*)(ws + SZ_X + 2 * SZ_W);
  bf16* WoT = (bf16*)(ws + SZ_X + 3 * SZ_W);
  bf16* Qb  = (bf16*)(ws + SZ_X + 4 * SZ_W);
  bf16* Kb  = (bf16*)(ws + SZ_X + 4 * SZ_W + SZ_X);
  bf16* VTb = (bf16*)(ws + SZ_X + 4 * SZ_W + 2 * SZ_X);

  k_convert_bf16<<<(BN * EMB + 255) / 256, 256, 0, stream>>>(x, xb, BN * EMB);
  k_transpose_bf16<<<(EMB * EMB + 255) / 256, 256, 0, stream>>>(Wq, WqT);
  k_transpose_bf16<<<(EMB * EMB + 255) / 256, 256, 0, stream>>>(Wk, WkT);
  k_transpose_bf16<<<(EMB * EMB + 255) / 256, 256, 0, stream>>>(Wv, WvT);
  k_transpose_bf16<<<(EMB * EMB + 255) / 256, 256, 0, stream>>>(Wo, WoT);

  // 3 matrices * 512 row-tiles * 12 col-groups = 18432 waves / 4 per block
  k_qkv_gemm<<<4608, 128, 0, stream>>>(xb, WqT, WkT, WvT, Qb, Kb, VTb);

  // 2 * 12 * 256 = 6144 waves / 4 per block
  k_attention<<<1536, 128, 0, stream>>>(Qb, Kb, VTb, xb);

  // 512 * 12 = 6144 waves / 4 per block
  k_out_gemm<<<1536, 128, 0, stream>>>(xb, WoT, bo, out);
}
